// CustomMHA_71150428225549
// MI455X (gfx1250) — matrix operand order repaired and checked by reading, not compiled, not run on hardware
//
#include <hip/hip_runtime.h>
#include <hip/hip_bf16.h>

typedef __attribute__((ext_vector_type(16))) _Float16 v16h;
typedef __attribute__((ext_vector_type(8)))  _Float16 v8h;
typedef __attribute__((ext_vector_type(8)))  float    v8f;
typedef int v4i __attribute__((vector_size(16)));

#define AS1 __attribute__((address_space(1)))
#define AS3 __attribute__((address_space(3)))

#define DMODEL 768
#define TSEQ   2048
#define NB     2
#define NH     12
#define DH     64
#define MROWS  (NB * TSEQ)          // 4096
#define OUT_ELEMS ((size_t)NB * TSEQ * DMODEL)       // 3,145,728
#define W_ELEMS   ((size_t)DMODEL * DMODEL)          // 589,824
#define HEAD_ELEMS ((size_t)NB * NH * TSEQ * DH)     // 3,145,728

// ---------------- WMMA fragment helpers (wave32) ----------------
// A-operand 16x32 f16: lane m = lane&15, kb = (lane>>4)*8,
//   halves = K[kb..kb+7], K[kb+16..kb+23]
__device__ __forceinline__ v16h frag_a(const _Float16* p, int ld) {
  int lane = threadIdx.x & 31;
  const _Float16* q = p + (size_t)(lane & 15) * ld + ((lane >> 4) << 3);
  v8h lo = *(const v8h*)q;
  v8h hi = *(const v8h*)(q + 16);
  v16h r;
#pragma unroll
  for (int i = 0; i < 8; ++i) { r[i] = lo[i]; r[i + 8] = hi[i]; }
  return r;
}

// B-operand 32x16 f16 loaded from B^T (row n contiguous in K):
//   lane n = lane&15, kb = (lane>>4)*8, halves = K[kb..kb+7], K[kb+16..kb+23]
__device__ __forceinline__ v16h frag_b(const _Float16* p, int ld) {
  int lane = threadIdx.x & 31;
  const _Float16* q = p + (size_t)(lane & 15) * ld + ((lane >> 4) << 3);
  v8h lo = *(const v8h*)q;
  v8h hi = *(const v8h*)(q + 16);
  v16h r;
#pragma unroll
  for (int i = 0; i < 8; ++i) { r[i] = lo[i]; r[i + 8] = hi[i]; }
  return r;
}

__device__ __forceinline__ v8f wmma16(v16h a, v16h b, v8f c) {
  return __builtin_amdgcn_wmma_f32_16x16x32_f16(false, a, false, b,
                                                (short)0, c, false, false);
}

__device__ __forceinline__ v8f zero8() {
  v8f z;
#pragma unroll
  for (int i = 0; i < 8; ++i) z[i] = 0.0f;
  return z;
}

// ---------------- CDNA5 async global->LDS copy ----------------
__device__ __forceinline__ void async_b128(const _Float16* g, _Float16* l) {
#if __has_builtin(__builtin_amdgcn_global_load_async_to_lds_b128)
  const void* gv = (const void*)g;
  void* lv = (void*)l;
  __builtin_amdgcn_global_load_async_to_lds_b128(
      (AS1 v4i*)gv, (AS3 v4i*)lv, 0, 0);
#else
  asm volatile("global_load_async_to_lds_b128 %0, %1, off"
               :: "v"((unsigned)(size_t)l), "v"(g) : "memory");
#endif
}

__device__ __forceinline__ void wait_async_le2() {
#if __has_builtin(__builtin_amdgcn_s_wait_asynccnt)
  __builtin_amdgcn_s_wait_asynccnt(2);
#else
  asm volatile("s_wait_asynccnt 0x2" ::: "memory");
#endif
}

__device__ __forceinline__ void wait_async_le0() {
#if __has_builtin(__builtin_amdgcn_s_wait_asynccnt)
  __builtin_amdgcn_s_wait_asynccnt(0);
#else
  asm volatile("s_wait_asynccnt 0x0" ::: "memory");
#endif
}

// ---------------- conversion kernels ----------------
__global__ void k_cvt_f16(const float* __restrict__ src,
                          _Float16* __restrict__ dst, int n) {
  int i = blockIdx.x * 256 + threadIdx.x;
  if (i < n) dst[i] = (_Float16)src[i];
}

__global__ void k_transpose_w(const float* __restrict__ W,
                              _Float16* __restrict__ WT) {
  int i = blockIdx.x * 256 + threadIdx.x;  // over 768*768
  if (i < DMODEL * DMODEL) {
    int k = i / DMODEL, n = i % DMODEL;
    WT[(size_t)n * DMODEL + k] = (_Float16)W[i];
  }
}

// ---------------- generic 4096x768x768 WMMA GEMM ----------------
// C = A(f16,[M,768]) * B^T(f16,[N=768,K=768]) + bias
// B tile (64 N x 64 K) staged in LDS via async copy, double buffered:
// all 8 waves in the block share it (8x traffic reduction).
// mode 1: store f16 head-major [b,h,t,d]   (Q,K)
// mode 2: store f16 head-major transposed [b,h,d,t]   (V)
// mode 3: store f32 row-major [M,768] to outf   (final projection)
__global__ __launch_bounds__(256) void k_gemm(const _Float16* __restrict__ A,
                                              const _Float16* __restrict__ BT,
                                              const float* __restrict__ bias,
                                              _Float16* __restrict__ outh,
                                              float* __restrict__ outf,
                                              int mode) {
  __shared__ _Float16 Bs[2][64 * 64];   // 2 x 8KB

  const int wave = threadIdx.x >> 5, lane = threadIdx.x & 31;
  const int m0 = blockIdx.x * 128 + wave * 16;
  const int n0 = blockIdx.y * 64;
  const int tid = threadIdx.x;

  // stage one 64(N) x 64(K) chunk of B^T into an LDS buffer (async)
  auto stage = [&](int k0, _Float16* buf) {
#pragma unroll
    for (int iss = 0; iss < 2; ++iss) {
      int id = tid + iss * 256;          // 512 16-byte tasks
      int row = id >> 3, ch = id & 7;
      async_b128(BT + (size_t)(n0 + row) * DMODEL + k0 + ch * 8,
                 buf + row * 64 + ch * 8);
    }
  };

  v8f acc[4];
#pragma unroll
  for (int nt = 0; nt < 4; ++nt) acc[nt] = zero8();

  stage(0, Bs[0]);
  const int NCHUNK = DMODEL / 64;        // 12
  for (int c = 0; c < NCHUNK; ++c) {
    const int cur = c & 1;
    if (c + 1 < NCHUNK) stage((c + 1) * 64, Bs[1 - cur]);

    // prefetch this chunk's A fragments from global while async copy flies
    v16h a0 = frag_a(A + (size_t)m0 * DMODEL + c * 64, DMODEL);
    v16h a1 = frag_a(A + (size_t)m0 * DMODEL + c * 64 + 32, DMODEL);

    if (c + 1 < NCHUNK) wait_async_le2(); else wait_async_le0();
    __syncthreads();

#pragma unroll
    for (int ks = 0; ks < 2; ++ks) {
      v16h a = ks ? a1 : a0;
#pragma unroll
      for (int nt = 0; nt < 4; ++nt) {
        v16h bf = frag_b(&Bs[cur][nt * 16 * 64 + ks * 32], 64);
        acc[nt] = wmma16(a, bf, acc[nt]);
      }
    }
    __syncthreads();   // protect Bs[cur] before it is refilled
  }

  const int hi = lane >> 4, nl = lane & 15;
#pragma unroll
  for (int nt = 0; nt < 4; ++nt) {
#pragma unroll
    for (int r = 0; r < 8; ++r) {
      int n = n0 + nt * 16 + nl;
      int m = m0 + r + 8 * hi;
      float val = acc[nt][r] + bias[n];
      if (mode == 1) {
        int b = m >> 11, t = m & (TSEQ - 1), h = n >> 6, d = n & 63;
        outh[(((size_t)(b * NH + h)) * TSEQ + t) * DH + d] = (_Float16)val;
      } else if (mode == 2) {
        int b = m >> 11, t = m & (TSEQ - 1), h = n >> 6, d = n & 63;
        outh[(((size_t)(b * NH + h)) * DH + d) * TSEQ + t] = (_Float16)val;
      } else {
        outf[(size_t)m * DMODEL + n] = val;
      }
    }
  }
}

// ---------------- fused attention: scores -> softmax*gate -> A out -> z ----
// grid: NB*NH*(TSEQ/16) blocks, 256 threads (8 waves).
// LDS: P[16][2048] f16 (64KB) + reductions + z staging.
__global__ __launch_bounds__(256) void k_attn(const _Float16* __restrict__ qh,
                                              const _Float16* __restrict__ kh,
                                              const _Float16* __restrict__ vT,
                                              const float* __restrict__ gates,
                                              float* __restrict__ Aout,
                                              _Float16* __restrict__ zh) {
  extern __shared__ char smem_raw[];
  _Float16* P   = (_Float16*)smem_raw;                       // 16*2048 f16
  float* red    = (float*)(smem_raw + 16 * TSEQ * 2);        // 256
  float* rowM   = red + 256;                                 // 16
  float* rowS   = rowM + 16;                                 // 16
  float* zstage = rowS + 16;                                 // 1024

  const int wave = threadIdx.x >> 5, lane = threadIdx.x & 31;
  const int hi = lane >> 4, nl = lane & 15;

  const int qt = blockIdx.x & (TSEQ / 16 - 1);
  const int bh = blockIdx.x >> 7;
  const int b = bh / NH, h = bh % NH;
  const int t0 = qt << 4;

  const _Float16* qbase = qh + (size_t)bh * TSEQ * DH;
  const _Float16* kbase = kh + (size_t)bh * TSEQ * DH;

  // ---- phase 1: scores tile (16 x 2048) via WMMA into LDS ----
  v16h qa0 = frag_a(qbase + (size_t)t0 * DH + 0, DH);
  v16h qa1 = frag_a(qbase + (size_t)t0 * DH + 32, DH);
  // prefetch first K tile; pipeline one tile ahead
  v16h kb0 = frag_b(kbase + (size_t)(wave * 256) * DH + 0, DH);
  v16h kb1 = frag_b(kbase + (size_t)(wave * 256) * DH + 32, DH);
#pragma unroll 4
  for (int i = 0; i < 16; ++i) {
    int s0 = wave * 256 + i * 16;
    v16h nb0 = kb0, nb1 = kb1;
    if (i + 1 < 16) {
      nb0 = frag_b(kbase + (size_t)(s0 + 16) * DH + 0, DH);
      nb1 = frag_b(kbase + (size_t)(s0 + 16) * DH + 32, DH);
    }
    v8f acc = zero8();
    acc = wmma16(qa0, kb0, acc);
    acc = wmma16(qa1, kb1, acc);
#pragma unroll
    for (int r = 0; r < 8; ++r) {
      int m = r + 8 * hi;
      P[m * TSEQ + s0 + nl] = (_Float16)(acc[r] * 0.125f);  // 1/sqrt(64)
    }
    kb0 = nb0; kb1 = nb1;
  }
  __syncthreads();

  // ---- phase 2: stable softmax per row, apply gate, emit A ----
  const float gate = gates[b * NH + h];
  const int row = threadIdx.x >> 4, j = threadIdx.x & 15;

  float mx = -1e30f;
  for (int s = j; s < TSEQ; s += 16)
    mx = fmaxf(mx, (float)P[row * TSEQ + s]);
  red[threadIdx.x] = mx;
  __syncthreads();
  if (j == 0) {
    float m2 = red[row * 16];
#pragma unroll
    for (int t2 = 1; t2 < 16; ++t2) m2 = fmaxf(m2, red[row * 16 + t2]);
    rowM[row] = m2;
  }
  __syncthreads();

  const float rm = rowM[row];
  float sum = 0.0f;
  for (int s = j; s < TSEQ; s += 16) {
    float p = __expf((float)P[row * TSEQ + s] - rm);
    P[row * TSEQ + s] = (_Float16)p;
    sum += p;
  }
  red[threadIdx.x] = sum;
  __syncthreads();
  if (j == 0) {
    float s2 = 0.0f;
#pragma unroll
    for (int t2 = 0; t2 < 16; ++t2) s2 += red[row * 16 + t2];
    rowS[row] = gate / s2;
  }
  __syncthreads();

  const float sc = rowS[row];
  float* Arow = Aout + ((size_t)bh * TSEQ + t0 + row) * TSEQ;
  for (int s = j; s < TSEQ; s += 16) {
    float a = (float)P[row * TSEQ + s] * sc;
    P[row * TSEQ + s] = (_Float16)a;   // gated probs feed z = P*V
    Arow[s] = a;                        // full attention matrix output
  }
  __syncthreads();

  // ---- phase 3: z = P * V (split-K across wave pairs) ----
  const int nt = wave & 3, ksplit = wave >> 2;
  const _Float16* vbase = vT + ((size_t)bh * DH + nt * 16) * TSEQ;
  const int kk0 = ksplit * 1024;
  v16h vb = frag_b(vbase + kk0, TSEQ);
  v8f z = zero8();
  for (int kk = kk0; kk < kk0 + 1024; kk += 32) {
    v16h vbn = vb;
    if (kk + 32 < kk0 + 1024) vbn = frag_b(vbase + kk + 32, TSEQ);
    v16h pa = frag_a(P + kk, TSEQ);        // probs from LDS (f16)
    z = wmma16(pa, vb, z);
    vb = vbn;
  }
  if (ksplit == 1) {
#pragma unroll
    for (int r = 0; r < 8; ++r) zstage[nt * 256 + r * 32 + lane] = z[r];
  }
  __syncthreads();
  if (ksplit == 0) {
#pragma unroll
    for (int r = 0; r < 8; ++r) {
      float v = z[r] + zstage[nt * 256 + r * 32 + lane];
      int t = t0 + r + 8 * hi;
      int col = h * DH + nt * 16 + nl;
      zh[((size_t)b * TSEQ + t) * DMODEL + col] = (_Float16)v;
    }
  }
}

// ---------------- host-side launcher ----------------
extern "C" void kernel_launch(void* const* d_in, const int* in_sizes, int n_in,
                              void* d_out, int out_size, void* d_ws, size_t ws_size,
                              hipStream_t stream) {
  const float* x     = (const float*)d_in[0];
  const float* gates = (const float*)d_in[1];
  const float* Wq    = (const float*)d_in[2];
  const float* bq    = (const float*)d_in[3];
  const float* Wk    = (const float*)d_in[4];
  const float* bk    = (const float*)d_in[5];
  const float* Wv    = (const float*)d_in[6];
  const float* bv    = (const float*)d_in[7];
  const float* Wo    = (const float*)d_in[8];
  const float* bo    = (const float*)d_in[9];
  float* out = (float*)d_out;

  // workspace carve-out (all f16, all 16B-aligned region sizes)
  _Float16* p = (_Float16*)d_ws;
  _Float16* xh  = p; p += OUT_ELEMS;     // [4096,768]
  _Float16* WqT = p; p += W_ELEMS;
  _Float16* WkT = p; p += W_ELEMS;
  _Float16* WvT = p; p += W_ELEMS;
  _Float16* WoT = p; p += W_ELEMS;
  _Float16* qhh = p; p += HEAD_ELEMS;    // [b,h,t,d]
  _Float16* khh = p; p += HEAD_ELEMS;    // [b,h,t,d]
  _Float16* vTT = p; p += HEAD_ELEMS;    // [b,h,d,t]
  _Float16* zh  = p; p += OUT_ELEMS;     // [4096,768]

  const int ne_x = (int)OUT_ELEMS;
  k_cvt_f16<<<(ne_x + 255) / 256, 256, 0, stream>>>(x, xh, ne_x);
  const int ne_w = (int)W_ELEMS;
  k_transpose_w<<<(ne_w + 255) / 256, 256, 0, stream>>>(Wq, WqT);
  k_transpose_w<<<(ne_w + 255) / 256, 256, 0, stream>>>(Wk, WkT);
  k_transpose_w<<<(ne_w + 255) / 256, 256, 0, stream>>>(Wv, WvT);
  k_transpose_w<<<(ne_w + 255) / 256, 256, 0, stream>>>(Wo, WoT);

  dim3 ggrid(MROWS / 128, DMODEL / 64);
  k_gemm<<<ggrid, 256, 0, stream>>>(xh, WqT, bq, qhh, nullptr, 1);
  k_gemm<<<ggrid, 256, 0, stream>>>(xh, WkT, bk, khh, nullptr, 1);
  k_gemm<<<ggrid, 256, 0, stream>>>(xh, WvT, bv, vTT, nullptr, 2);

  const size_t smem = (size_t)16 * TSEQ * 2           // P (f16)
                    + (256 + 16 + 16 + 1024) * 4;     // red/rowM/rowS/zstage
  k_attn<<<NB * NH * (TSEQ / 16), 256, smem, stream>>>(
      qhh, khh, vTT, gates, out + OUT_ELEMS, zh);

  k_gemm<<<ggrid, 256, 0, stream>>>(zh, WoT, bo, nullptr, out, 3);
}
